// OptimizedSequentialMNISTModel_86474871538180
// MI455X (gfx1250) — matrix operand order repaired and checked by reading, not compiled, not run on hardware
//
#include <hip/hip_runtime.h>

typedef __attribute__((ext_vector_type(16))) _Float16 v16h;
typedef __attribute__((ext_vector_type(8)))  _Float16 v8h;
typedef __attribute__((ext_vector_type(8)))  float    v8f;

#define WMMA(a, b, c) __builtin_amdgcn_wmma_f32_16x16x32_f16( \
    false, (a), false, (b), (short)0, (c), false, false)

constexpr int T_STEPS  = 98;   // ceil(784 / 8)
constexpr int XP16     = 104;  // f16 pitch for x staging
constexpr int WAVES    = 4;    // waves per block
constexpr int C1_PITCH = 32;   // comb1: [x | h1(24) | pad]       16x32 f16
constexpr int NB_PITCH = 96;   // nbs (shared by cell1 & cell2)   16x96 f16
constexpr int C2_PITCH = 96;   // comb2: [h1(24) | h2(48) | pad]  16x96 f16
constexpr int HALFS_PER_WAVE = 16 * (C1_PITCH + NB_PITCH + C2_PITCH); // 3584
constexpr int WB2_FRAGS = 18;  // 3 K-chunks x 6 N-tiles
constexpr int WB2_HALFS = WB2_FRAGS * 32 * 16; // per-frag: 32 lanes x 16 halves

__device__ __forceinline__ float fast_tanh(float x) {
#if __has_builtin(__builtin_amdgcn_tanhf)
    return __builtin_amdgcn_tanhf(x);   // v_tanh_f32 (gfx1250 TRANS op)
#else
    // tanh(x) = 1 - 2/(e^{2x}+1); exp2-based, saturates to +/-1 correctly
    float t = __builtin_amdgcn_exp2f(x * 2.8853900817779268f); // 2*log2(e)
    return 1.0f - 2.0f * __builtin_amdgcn_rcpf(t + 1.0f);
#endif
}

__device__ __forceinline__ v8f bcast8(float v) {
    v8f r;
#pragma unroll
    for (int i = 0; i < 8; ++i) r[i] = v;
    return r;
}

__device__ __forceinline__ float ld_bias(const float* b, int N, int idx) {
    return (idx < N) ? b[idx] : 0.0f;
}

// Build a 32x16 f16 B-fragment of W^T (i.e. B[k][n] = W[n][k]), W row-major (N x K).
// Lane l holds column n = nTile*16 + (l&15), K = kBase + (l<16 ? 0..7,16..23 : 8..15,24..31).
__device__ __forceinline__ v16h load_bfrag(const float* __restrict__ W, int N, int K,
                                           int ldk, int nTile, int kBase, int lane) {
    int n  = nTile * 16 + (lane & 15);
    int k0 = kBase + ((lane < 16) ? 0 : 8);
    v16h b;
#pragma unroll
    for (int i = 0; i < 16; ++i) {
        int   k = k0 + (i < 8 ? i : i + 8);
        float v = (n < N && k < K) ? W[n * ldk + k] : 0.0f;
        b[i] = (_Float16)v;
    }
    return b;
}

// Load a 16x32 f16 A-fragment from row-major LDS buffer (pitch in halves, rows 16B aligned).
__device__ __forceinline__ v16h load_afrag(const _Float16* __restrict__ buf,
                                           int pitch, int kBase, int lane) {
    int row = lane & 15;
    int off = kBase + ((lane >> 4) ? 8 : 0);
    v8h lo = *(const v8h*)(buf + row * pitch + off);
    v8h hi = *(const v8h*)(buf + row * pitch + off + 16);
    return __builtin_shufflevector(lo, hi, 0, 1, 2, 3, 4, 5, 6, 7,
                                   8, 9, 10, 11, 12, 13, 14, 15);
}

// Store a 16x16 f32 C/D tile as f16 into row-major LDS buffer.
// C/D layout: VGPR r, lane l -> row r + 8*(l>>4), col l&15. colLimit masks invalid cols.
__device__ __forceinline__ void store_dtile(_Float16* __restrict__ buf, int pitch,
                                            int colBase, const v8f d, int lane,
                                            int colLimit) {
    int c = lane & 15;
    if (c < colLimit) {
        int col = colBase + c;
        int rb  = (lane >> 4) * 8;
#pragma unroll
        for (int r = 0; r < 8; ++r) buf[(rb + r) * pitch + col] = (_Float16)d[r];
    }
}

__global__ __launch_bounds__(128, 1) void branch_rnn_kernel(
    const float* __restrict__ x,
    const float* __restrict__ Wb1, const float* __restrict__ bb1,
    const float* __restrict__ Wg1, const float* __restrict__ bg1,
    const float* __restrict__ Wb2, const float* __restrict__ bb2,
    const float* __restrict__ Wg2, const float* __restrict__ bg2,
    const float* __restrict__ Wc1, const float* __restrict__ bc1,
    const float* __restrict__ Wc2, const float* __restrict__ bc2,
    float* __restrict__ out) {
    __shared__ __align__(16) _Float16 smem[WAVES * HALFS_PER_WAVE];  // 28,672 B
    __shared__ __align__(16) _Float16 xsm[WAVES * 16 * XP16];        // 13,312 B
    __shared__ __align__(16) _Float16 wfrag[WB2_HALFS];              // 18,432 B (block-shared)

    const int wave = threadIdx.x >> 5;
    const int lane = threadIdx.x & 31;
    const int col  = lane & 15;
    const int rowBase = (blockIdx.x * WAVES + wave) * 16;

    _Float16* c1 = smem + wave * HALFS_PER_WAVE;
    _Float16* nb = c1 + 16 * C1_PITCH;   // shared nbs buffer (cell1: cols 0..47, cell2: 0..95)
    _Float16* c2 = nb + 16 * NB_PITCH;
    _Float16* xb = xsm + wave * 16 * XP16;

    // Zero wave-private staging (h=0 init + static zero pads).
    for (int i = lane; i < HALFS_PER_WAVE / 2; i += 32) ((int*)c1)[i] = 0;

    // Prefetch this wave's subsampled x into LDS as f16: xb[row][t] = x[rowBase+row][8t].
    for (int i = lane; i < 16 * T_STEPS; i += 32) {
        int r = i / T_STEPS;
        int t = i - r * T_STEPS;
        xb[r * XP16 + t] = (_Float16)x[(size_t)(rowBase + r) * 784 + 8 * t];
    }

    // Wave 0 fills the block-shared, pre-swizzled Wb2 B-fragments (largest weight set):
    // layout: [frag = kc*6+nt][lane][16 halves] -> each in-loop read is one aligned 32B access.
    if (wave == 0) {
#pragma unroll
        for (int kc = 0; kc < 3; ++kc)
#pragma unroll
            for (int nt = 0; nt < 6; ++nt) {
                v16h f = load_bfrag(Wb2, 96, 72, 72, nt, kc * 32, lane);
                *(v16h*)(wfrag + (kc * 6 + nt) * 512 + lane * 16) = f;
            }
    }

    // ---- Remaining B fragments stay register-resident across the scan ----
    v16h fWb1[3];
#pragma unroll
    for (int nt = 0; nt < 3; ++nt) fWb1[nt] = load_bfrag(Wb1, 48, 25, 25, nt, 0, lane);
    v16h fWg1[2][2];
#pragma unroll
    for (int kc = 0; kc < 2; ++kc)
#pragma unroll
        for (int nt = 0; nt < 2; ++nt)
            fWg1[kc][nt] = load_bfrag(Wg1, 24, 48, 48, nt, kc * 32, lane);
    v16h fWg2[3][3];
#pragma unroll
    for (int kc = 0; kc < 3; ++kc)
#pragma unroll
        for (int nt = 0; nt < 3; ++nt)
            fWg2[kc][nt] = load_bfrag(Wg2, 48, 96, 96, nt, kc * 32, lane);

    float bB1[3], bG1[2], bB2[6], bG2[3];
#pragma unroll
    for (int nt = 0; nt < 3; ++nt) bB1[nt] = ld_bias(bb1, 48, nt * 16 + col);
#pragma unroll
    for (int nt = 0; nt < 2; ++nt) bG1[nt] = ld_bias(bg1, 24, nt * 16 + col);
#pragma unroll
    for (int nt = 0; nt < 6; ++nt) bB2[nt] = ld_bias(bb2, 96, nt * 16 + col);
#pragma unroll
    for (int nt = 0; nt < 3; ++nt) bG2[nt] = ld_bias(bg2, 48, nt * 16 + col);

    // Branch-blend weight: col parity == lane parity (col = tile*16 + (lane&15), 16 even)
    const float wS = (lane & 1) ? 0.6f : 0.4f; // applied to old state (bw)
    const float wO = 1.0f - wS;                // applied to new branch output

    v8f s1[3], s2[6]; // branch states, kept in C/D layout (f32)
#pragma unroll
    for (int nt = 0; nt < 3; ++nt) s1[nt] = bcast8(0.0f);
#pragma unroll
    for (int nt = 0; nt < 6; ++nt) s2[nt] = bcast8(0.0f);

    __syncthreads(); // wfrag visible to all waves (only cross-wave sync in the kernel)

    // ======================= scan over T (serial) =======================
    for (int t = 0; t < T_STEPS; ++t) {
        // comb1 col 0 <- x_t (cols 1..24 hold previous h1, 25..31 stay zero)
        if (lane < 16) c1[lane * C1_PITCH] = xb[lane * XP16 + t];

        // ---- cell1 GEMM1: comb1(16x25) @ Wb1^T(25x48), blend into s1 ----
        v16h a0 = load_afrag(c1, C1_PITCH, 0, lane);
#pragma unroll
        for (int nt = 0; nt < 3; ++nt) {
            v8f acc = bcast8(bB1[nt]);
            acc = WMMA(a0, fWb1[nt], acc);
#pragma unroll
            for (int r = 0; r < 8; ++r) s1[nt][r] = wS * s1[nt][r] + wO * acc[r];
            store_dtile(nb, NB_PITCH, nt * 16, s1[nt], lane, 16);
        }

        // ---- cell1 GEMM2: nbs1(16x48) @ Wg1^T(48x24), tanh -> h1 ----
        // (cols 48..63 of nb hold stale cell2 data; safe: B rows k>=48 are zero)
        v16h p0 = load_afrag(nb, NB_PITCH, 0, lane);
        v16h p1 = load_afrag(nb, NB_PITCH, 32, lane);
#pragma unroll
        for (int nt = 0; nt < 2; ++nt) {
            v8f acc = bcast8(bG1[nt]);
            acc = WMMA(p0, fWg1[0][nt], acc);
            acc = WMMA(p1, fWg1[1][nt], acc);
#pragma unroll
            for (int r = 0; r < 8; ++r) acc[r] = fast_tanh(acc[r]);
            int lim = nt ? 8 : 16;                              // h1 has 24 cols
            store_dtile(c1, C1_PITCH, 1 + nt * 16, acc, lane, lim); // -> comb1[1..24]
            store_dtile(c2, C2_PITCH, nt * 16, acc, lane, lim);     // -> comb2[0..23]
        }

        // ---- cell2 GEMM1: comb2(16x72) @ Wb2^T(72x96), blend into s2 ----
        v16h q0 = load_afrag(c2, C2_PITCH, 0, lane);
        v16h q1 = load_afrag(c2, C2_PITCH, 32, lane);
        v16h q2 = load_afrag(c2, C2_PITCH, 64, lane);
#pragma unroll
        for (int nt = 0; nt < 6; ++nt) {
            v16h b0 = *(const v16h*)(wfrag + (0 * 6 + nt) * 512 + lane * 16);
            v16h b1 = *(const v16h*)(wfrag + (1 * 6 + nt) * 512 + lane * 16);
            v16h b2 = *(const v16h*)(wfrag + (2 * 6 + nt) * 512 + lane * 16);
            v8f acc = bcast8(bB2[nt]);
            acc = WMMA(q0, b0, acc);
            acc = WMMA(q1, b1, acc);
            acc = WMMA(q2, b2, acc);
#pragma unroll
            for (int r = 0; r < 8; ++r) s2[nt][r] = wS * s2[nt][r] + wO * acc[r];
            store_dtile(nb, NB_PITCH, nt * 16, s2[nt], lane, 16);
        }

        // ---- cell2 GEMM2: nbs2(16x96) @ Wg2^T(96x48), tanh -> h2 ----
        v16h r0 = load_afrag(nb, NB_PITCH, 0, lane);
        v16h r1 = load_afrag(nb, NB_PITCH, 32, lane);
        v16h r2 = load_afrag(nb, NB_PITCH, 64, lane);
#pragma unroll
        for (int nt = 0; nt < 3; ++nt) {
            v8f acc = bcast8(bG2[nt]);
            acc = WMMA(r0, fWg2[0][nt], acc);
            acc = WMMA(r1, fWg2[1][nt], acc);
            acc = WMMA(r2, fWg2[2][nt], acc);
#pragma unroll
            for (int r = 0; r < 8; ++r) acc[r] = fast_tanh(acc[r]);
            store_dtile(c2, C2_PITCH, 24 + nt * 16, acc, lane, 16); // h2 -> comb2[24..71]
        }
    }

    // ======================= classifier head =======================
    v16h fWc1[2][2];
#pragma unroll
    for (int kc = 0; kc < 2; ++kc)
#pragma unroll
        for (int nt = 0; nt < 2; ++nt)
            fWc1[kc][nt] = load_bfrag(Wc1, 32, 48, 48, nt, kc * 32, lane);
    float bC1[2];
#pragma unroll
    for (int nt = 0; nt < 2; ++nt) bC1[nt] = ld_bias(bc1, 32, nt * 16 + col);

    // hidden = relu(h2 @ Wc1^T + bc1); h2 sits in comb2 cols 24..71 (pad cols are zero)
    v16h h0 = load_afrag(c2 + 24, C2_PITCH, 0, lane);
    v16h h1 = load_afrag(c2 + 24, C2_PITCH, 32, lane);
#pragma unroll
    for (int nt = 0; nt < 2; ++nt) {
        v8f acc = bcast8(bC1[nt]);
        acc = WMMA(h0, fWc1[0][nt], acc);
        acc = WMMA(h1, fWc1[1][nt], acc);
#pragma unroll
        for (int r = 0; r < 8; ++r) acc[r] = fmaxf(acc[r], 0.0f);
        store_dtile(c1, C1_PITCH, nt * 16, acc, lane, 16); // hidden (16x32) -> c1
    }

    // logits = hidden @ Wc2^T + bc2
    v16h fWc2 = load_bfrag(Wc2, 10, 32, 32, 0, 0, lane);
    v16h hh   = load_afrag(c1, C1_PITCH, 0, lane);
    v8f  lg   = bcast8(ld_bias(bc2, 10, col));
    lg = WMMA(hh, fWc2, lg);

    if (col < 10) {
        int rb = (lane >> 4) * 8;
#pragma unroll
        for (int r = 0; r < 8; ++r)
            out[(size_t)(rowBase + rb + r) * 10 + col] = lg[r];
    }
}

extern "C" void kernel_launch(void* const* d_in, const int* in_sizes, int n_in,
                              void* d_out, int out_size, void* d_ws, size_t ws_size,
                              hipStream_t stream) {
    (void)n_in; (void)d_ws; (void)ws_size; (void)out_size;
    const float* x   = (const float*)d_in[0];
    const float* Wb1 = (const float*)d_in[1];
    const float* bb1 = (const float*)d_in[2];
    const float* Wg1 = (const float*)d_in[3];
    const float* bg1 = (const float*)d_in[4];
    const float* Wb2 = (const float*)d_in[5];
    const float* bb2 = (const float*)d_in[6];
    const float* Wg2 = (const float*)d_in[7];
    const float* bg2 = (const float*)d_in[8];
    const float* Wc1 = (const float*)d_in[9];
    const float* bc1 = (const float*)d_in[10];
    const float* Wc2 = (const float*)d_in[11];
    const float* bc2 = (const float*)d_in[12];
    float* out = (float*)d_out;

    int B = in_sizes[0] / 784;           // 65536
    int rowsPerBlock = WAVES * 16;       // 64 rows per block (4 waves x 16)
    int grid = (B + rowsPerBlock - 1) / rowsPerBlock;

    branch_rnn_kernel<<<dim3(grid), dim3(WAVES * 32), 0, stream>>>(
        x, Wb1, bb1, Wg1, bg1, Wb2, bb2, Wg2, bg2, Wc1, bc1, Wc2, bc2, out);
}
